// BinResNetConv2d_21912923144908
// MI455X (gfx1250) — matrix operand order repaired and checked by reading, not compiled, not run on hardware
//
#include <hip/hip_runtime.h>
#include <hip/hip_bf16.h>
#include <stdint.h>

typedef _Float16 half_t;
typedef __attribute__((ext_vector_type(16))) _Float16 v16h;
typedef __attribute__((ext_vector_type(8)))  _Float16 v8h;
typedef __attribute__((ext_vector_type(8)))  float    v8f;

// Problem constants
#define NB   32          // batch
#define C    256         // channels in/out
#define H    56
#define W    56
#define HP   58          // padded
#define WP   58
#define KDIM 2304        // C*9
#define SPAT 3136        // H*W per image

// Workspace layout (bytes)
#define XB_ELEMS   ((size_t)NB * HP * WP * C)          // 27,557,888 f16
#define XB_BYTES   (XB_ELEMS * 2)                      // 55,115,776
#define W16_ELEMS  ((size_t)C * KDIM)                  // 589,824 f16

// ---------------------------------------------------------------------------
// Prep 1: BatchNorm (inference) + sign -> f16 padded NHWC buffer
// xb[((n*58 + hp)*58 + wp)*256 + c], zero halo (conv pads binarized input w/ 0)
// ---------------------------------------------------------------------------
__global__ __launch_bounds__(256)
void bn_bin_kernel(const float* __restrict__ x,
                   const float* __restrict__ gamma,
                   const float* __restrict__ beta,
                   const float* __restrict__ mean,
                   const float* __restrict__ var,
                   half_t* __restrict__ xb) {
  const uint32_t idx = blockIdx.x * blockDim.x + threadIdx.x;
  if (idx >= (uint32_t)XB_ELEMS) return;
  const int c  = idx & 255;
  uint32_t t   = idx >> 8;
  const int wp = t % WP;  t /= WP;
  const int hp = t % HP;
  const int n  = t / HP;
  half_t v = (half_t)0.0f;
  if (hp >= 1 && hp <= H && wp >= 1 && wp <= W) {
    const float inv_std = rsqrtf(var[c] + 1e-5f);
    const float scale   = gamma[c] * inv_std;
    const float shift   = beta[c] - mean[c] * scale;
    const float xv = x[(((uint32_t)(n * C + c) * H) + (hp - 1)) * W + (wp - 1)];
    const float xn = fmaf(xv, scale, shift);
    v = (half_t)((xn > 0.0f) ? 1.0f : ((xn < 0.0f) ? -1.0f : 0.0f));
  }
  xb[idx] = v;
}

// ---------------------------------------------------------------------------
// Prep 2: weights OIHW f32 -> f16 [oc][k], k = (kh*3+kw)*256 + cin
// ---------------------------------------------------------------------------
__global__ __launch_bounds__(256)
void wprep_kernel(const float* __restrict__ w, half_t* __restrict__ w16) {
  const uint32_t idx = blockIdx.x * blockDim.x + threadIdx.x;
  if (idx >= (uint32_t)W16_ELEMS) return;
  const int k   = idx % KDIM;
  const int oc  = idx / KDIM;
  const int cin = k & 255;
  const int kk  = k >> 8;          // kh*3+kw
  const int kh  = kk / 3;
  const int kw  = kk % 3;
  w16[idx] = (half_t)w[(((uint32_t)(oc * C + cin) * 3) + kh) * 3 + kw];
}

// ---------------------------------------------------------------------------
// Main: implicit-GEMM binarized conv via v_wmma_f32_16x16x32_f16.
// Each wave: 2 oc-tiles x 4 spatial-tiles = 32 ocs x 64 pixels.
// K loop = 9 taps x 8 cin-chunks; 8 WMMAs per step, 12 b128 loads per step.
// ---------------------------------------------------------------------------
__global__ __launch_bounds__(256)
void binconv_wmma_kernel(const half_t* __restrict__ xb,
                         const half_t* __restrict__ w16,
                         float* __restrict__ out) {
  const int lane = threadIdx.x & 31;
  const int wid  = blockIdx.x * 8 + (threadIdx.x >> 5);

  const int ocg = wid & 7;            // 8 groups of 32 oc
  int t         = wid >> 3;
  const int sp  = t % 49;             // 49 groups of 64 pixels per image
  const int img = t / 49;

  const int ocbase = ocg * 32;
  const int p0     = sp * 64;
  const int laneN  = lane & 15;
  const int laneM  = lane & 15;
  const int hi     = lane >> 4;       // 0 or 1: which K-half this lane carries

  // B operand base offsets (element units) for the four spatial tiles,
  // excluding the (kh,kw) tap shift and the cin-chunk offset.
  uint32_t bBase[4];
#pragma unroll
  for (int nt = 0; nt < 4; ++nt) {
    const int p = p0 + nt * 16 + laneN;
    const int h = p / W;
    const int w = p % W;
    // padded coords for tap (kh,kw): row = h+kh, col = w+kw (added later)
    bBase[nt] = ((uint32_t)(img * HP + h) * WP + w) * (uint32_t)C
              + (uint32_t)hi * 16u;   // lanes>=16 carry K = c0+16..c0+31
  }

  // A operand base offsets: lanes 0-15 carry K {k0..k0+15},
  // lanes 16-31 carry K {k0+16..k0+31}.
  const uint32_t aBase0 = (uint32_t)(ocbase + laneM) * (uint32_t)KDIM
                        + (uint32_t)hi * 16u;
  const uint32_t aBase1 = aBase0 + 16u * (uint32_t)KDIM;

  v8f acc[2][4];
#pragma unroll
  for (int mt = 0; mt < 2; ++mt)
#pragma unroll
    for (int nt = 0; nt < 4; ++nt)
      acc[mt][nt] = (v8f){};

  for (int kk = 0; kk < 9; ++kk) {
    const int kh = kk / 3;
    const int kw = kk % 3;
    const uint32_t tapOff = ((uint32_t)kh * WP + (uint32_t)kw) * (uint32_t)C;
    const uint32_t kbase  = (uint32_t)kk * 256u;
#pragma unroll
    for (int cc = 0; cc < 8; ++cc) {
      const uint32_t c0 = (uint32_t)cc * 32u;
      const uint32_t k0 = kbase + c0;

      // --- A tiles: two contiguous 16B chunks per lane ---
      v16h a0, a1;
      const half_t* pa0 = w16 + aBase0 + k0;
      ((v8h*)&a0)[0] = *(const v8h*)(pa0);
      ((v8h*)&a0)[1] = *(const v8h*)(pa0 + 8);
      const half_t* pa1 = w16 + aBase1 + k0;
      ((v8h*)&a1)[0] = *(const v8h*)(pa1);
      ((v8h*)&a1)[1] = *(const v8h*)(pa1 + 8);

      // --- B tiles: one contiguous 32B chunk per lane (NHWC, cin-fastest) ---
      v16h b[4];
#pragma unroll
      for (int nt = 0; nt < 4; ++nt)
        b[nt] = *(const v16h*)(xb + bBase[nt] + tapOff + c0);

#pragma unroll
      for (int nt = 0; nt < 4; ++nt) {
        acc[0][nt] = __builtin_amdgcn_wmma_f32_16x16x32_f16(
            false, a0, false, b[nt], (short)0, acc[0][nt], false, false);
        acc[1][nt] = __builtin_amdgcn_wmma_f32_16x16x32_f16(
            false, a1, false, b[nt], (short)0, acc[1][nt], false, false);
      }
    }
  }

  // Epilogue: ReLU + store. D layout: VGPR r -> lanes 0-15: (M=r, N=lane),
  // lanes 16-31: (M=8+r, N=lane-16). Output NCHW f32.
  const int mo = hi * 8;
#pragma unroll
  for (int mt = 0; mt < 2; ++mt) {
#pragma unroll
    for (int nt = 0; nt < 4; ++nt) {
      const v8f accv = acc[mt][nt];
      const int p = p0 + nt * 16 + laneN;
      float* outp = out + ((uint64_t)(img * C + ocbase + mt * 16 + mo) * SPAT) + p;
#pragma unroll
      for (int r = 0; r < 8; ++r) {
        outp[(uint64_t)r * SPAT] = fmaxf(accv[r], 0.0f);
      }
    }
  }
}

// ---------------------------------------------------------------------------
extern "C" void kernel_launch(void* const* d_in, const int* in_sizes, int n_in,
                              void* d_out, int out_size, void* d_ws, size_t ws_size,
                              hipStream_t stream) {
  const float* x     = (const float*)d_in[0];
  const float* w     = (const float*)d_in[1];
  const float* gamma = (const float*)d_in[2];
  const float* beta  = (const float*)d_in[3];
  const float* rmean = (const float*)d_in[4];
  const float* rvar  = (const float*)d_in[5];
  float* out         = (float*)d_out;

  half_t* xb  = (half_t*)d_ws;
  half_t* w16 = (half_t*)((char*)d_ws + XB_BYTES);

  // Prep 1: BN + binarize -> padded NHWC f16
  {
    const uint32_t total  = (uint32_t)XB_ELEMS;
    const uint32_t blocks = (total + 255u) / 256u;
    bn_bin_kernel<<<blocks, 256, 0, stream>>>(x, gamma, beta, rmean, rvar, xb);
  }
  // Prep 2: weights -> f16 [oc][k]
  {
    const uint32_t total  = (uint32_t)W16_ELEMS;
    const uint32_t blocks = (total + 255u) / 256u;
    wprep_kernel<<<blocks, 256, 0, stream>>>(w, w16);
  }
  // Main WMMA conv: 8 oc-groups * 49 pixel-groups * 32 images = 12544 waves,
  // 8 waves (256 threads) per block -> 1568 blocks.
  {
    binconv_wmma_kernel<<<1568, 256, 0, stream>>>(xb, w16, out);
  }
}
